// MyExpert_39539468927000
// MI455X (gfx1250) — matrix operand order repaired and checked by reading, not compiled, not run on hardware
//
#include <hip/hip_runtime.h>
#include <cstdint>

// Problem constants (from reference)
#define E_   8
#define D_   1024
#define H_   4096
#define N_   4096

// Tiling
#define BM   128
#define BN   128
#define BK   32          // K per LDS stage (matches f16 WMMA K=32)
#define LSTR 48          // f16 LDS row stride (BK + 16 pad; keeps 16B-aligned chunks)

typedef __attribute__((ext_vector_type(16))) _Float16 v16h;
typedef __attribute__((ext_vector_type(8)))  float    v8f;

union F16Frag { v16h v; uint4 q[2]; _Float16 h[16]; };
union H2U     { _Float16 h; unsigned short u; };

// D = A*B + C three times: hi*hi + lo*hi + hi*lo  (2-term f16 split of fp32)
__device__ __forceinline__ v8f wmma3(v16h ah, v16h al, v16h bh, v16h bl, v8f c) {
  c = __builtin_amdgcn_wmma_f32_16x16x32_f16(false, ah, false, bh, (short)0, c, false, false);
  c = __builtin_amdgcn_wmma_f32_16x16x32_f16(false, al, false, bh, (short)0, c, false, false);
  c = __builtin_amdgcn_wmma_f32_16x16x32_f16(false, ah, false, bl, (short)0, c, false, false);
  return c;
}

// Split 4 fp32 -> (hi,lo) f16 and store 8B each into the two LDS planes.
__device__ __forceinline__ void store_split(_Float16* hi, _Float16* lo, float4 v) {
  float f[4] = {v.x, v.y, v.z, v.w};
  H2U hs[4], ls[4];
#pragma unroll
  for (int j = 0; j < 4; ++j) {
    hs[j].h = (_Float16)f[j];
    ls[j].h = (_Float16)(f[j] - (float)hs[j].h);
  }
  uint2 hp, lp;
  hp.x = (uint32_t)hs[0].u | ((uint32_t)hs[1].u << 16);
  hp.y = (uint32_t)hs[2].u | ((uint32_t)hs[3].u << 16);
  lp.x = (uint32_t)ls[0].u | ((uint32_t)ls[1].u << 16);
  lp.y = (uint32_t)ls[2].u | ((uint32_t)ls[3].u << 16);
  *reinterpret_cast<uint2*>(hi) = hp;
  *reinterpret_cast<uint2*>(lo) = lp;
}

// Per-thread slot geometry for staging a 128x32 fp32 tile with 256 threads:
// 1024 float4 slots, 4 per thread; slot -> (row, col4)
__device__ __forceinline__ void slot_rc(int tid, int it, int& r, int& c) {
  const int idx = tid + it * 256;
  r = idx >> 3;            // 8 float4 per 32-elem row
  c = (idx & 7) << 2;
}

// ---------------------------------------------------------------------------
// Kernel 1: h = gelu(X @ W1[e]^T + b1[e]) for the expert group owning each row.
// Writes h as packed (hi|lo<<16) f16 pairs to workspace.
// grid = (H/BN, N/BM, 2 batches), block = 256 threads (8 waves).
// ---------------------------------------------------------------------------
__global__ __launch_bounds__(256)
void moe_gemm1(const float* __restrict__ Xx, const float* __restrict__ Xy,
               const float* __restrict__ W1, const float* __restrict__ B1,
               const int*   __restrict__ counts,
               uint32_t*    __restrict__ Hbuf)
{
  __shared__ _Float16 As[2][BM][LSTR];   // [prec: 0=hi,1=lo][row m][k]
  __shared__ _Float16 Bs[2][BN][LSTR];   // [prec][row n][k]

  const int batch = blockIdx.z;
  const float* X  = batch ? Xy : Xx;
  uint32_t* Hout  = Hbuf + (size_t)batch * N_ * H_;

  const int m0 = blockIdx.y * BM;
  const int n0 = blockIdx.x * BN;

  int cum[E_ + 1];
  cum[0] = 0;
#pragma unroll
  for (int e = 0; e < E_; ++e) cum[e + 1] = cum[e] + counts[e];

  const int tid  = threadIdx.x;
  const int lane = tid & 31;
  const int wave = tid >> 5;
  const int half = lane >> 4;
  const int l16  = lane & 15;
  const int rowW = (wave >> 1) * 32;   // wave M offset (4 waves in M)
  const int colW = (wave & 1)  * 64;   // wave N offset (2 waves in N)

  v8f acc[2][4];
#pragma unroll
  for (int i = 0; i < 2; ++i)
#pragma unroll
    for (int j = 0; j < 4; ++j) acc[i][j] = v8f{};

  for (int e = 0; e < E_; ++e) {
    const int rs = max(m0, cum[e]);
    const int re = min(m0 + BM, cum[e + 1]);
    if (rs >= re) continue;                         // block-uniform
    const float* We = W1 + (size_t)e * H_ * D_;

    // ---- register-stage prefetch of tile k0 = 0 ----
    float4 ra[4], rb[4];
#pragma unroll
    for (int it = 0; it < 4; ++it) {
      int r, c; slot_rc(tid, it, r, c);
      const int grow = m0 + r;
      float4 av = make_float4(0.f, 0.f, 0.f, 0.f);
      if (grow >= rs && grow < re)
        av = *reinterpret_cast<const float4*>(X + (size_t)grow * D_ + c);
      ra[it] = av;
      rb[it] = *reinterpret_cast<const float4*>(We + (size_t)(n0 + r) * D_ + c);
    }

    for (int k0 = 0; k0 < D_; k0 += BK) {
      __syncthreads();   // previous compute done reading LDS
      // ---- drain prefetch registers into LDS (fp32 -> hi/lo f16 split) ----
#pragma unroll
      for (int it = 0; it < 4; ++it) {
        int r, c; slot_rc(tid, it, r, c);
        store_split(&As[0][r][c], &As[1][r][c], ra[it]);
        store_split(&Bs[0][r][c], &Bs[1][r][c], rb[it]);
      }
      __syncthreads();

      // ---- issue next tile's global loads; wait lands at next LDS store ----
      if (k0 + BK < D_) {
        const int kn = k0 + BK;
#pragma unroll
        for (int it = 0; it < 4; ++it) {
          int r, c; slot_rc(tid, it, r, c);
          const int grow = m0 + r;
          float4 av = make_float4(0.f, 0.f, 0.f, 0.f);
          if (grow >= rs && grow < re)
            av = *reinterpret_cast<const float4*>(X + (size_t)grow * D_ + kn + c);
          ra[it] = av;
          rb[it] = *reinterpret_cast<const float4*>(We + (size_t)(n0 + r) * D_ + kn + c);
        }
      }

      // ---- A fragments (16x32 f16 layout: lane<16 -> K{0..7,16..23}) ----
      F16Frag aH[2], aL[2];
#pragma unroll
      for (int mf = 0; mf < 2; ++mf) {
        const _Float16* ah = &As[0][rowW + mf * 16 + l16][0];
        const _Float16* al = &As[1][rowW + mf * 16 + l16][0];
        aH[mf].q[0] = *reinterpret_cast<const uint4*>(ah + 8 * half);
        aH[mf].q[1] = *reinterpret_cast<const uint4*>(ah + 16 + 8 * half);
        aL[mf].q[0] = *reinterpret_cast<const uint4*>(al + 8 * half);
        aL[mf].q[1] = *reinterpret_cast<const uint4*>(al + 16 + 8 * half);
      }
      // ---- B fragments (32x16 f16 layout: lane<16 -> K{0..7,16..23}) ----
#pragma unroll
      for (int nf = 0; nf < 4; ++nf) {
        F16Frag bH, bL;
        const _Float16* bh = &Bs[0][colW + nf * 16 + l16][0];
        const _Float16* bl = &Bs[1][colW + nf * 16 + l16][0];
        bH.q[0] = *reinterpret_cast<const uint4*>(bh + 8 * half);
        bH.q[1] = *reinterpret_cast<const uint4*>(bh + 16 + 8 * half);
        bL.q[0] = *reinterpret_cast<const uint4*>(bl + 8 * half);
        bL.q[1] = *reinterpret_cast<const uint4*>(bl + 16 + 8 * half);
#pragma unroll
        for (int mf = 0; mf < 2; ++mf)
          acc[mf][nf] = wmma3(aH[mf].v, aL[mf].v, bH.v, bL.v, acc[mf][nf]);
      }
    }
    __syncthreads();   // protect LDS before next expert pass re-stages
  }

  // ---- epilogue: +bias, exact-erf GELU, split to f16 pair, store ----
#pragma unroll
  for (int mf = 0; mf < 2; ++mf) {
#pragma unroll
    for (int r = 0; r < 8; ++r) {
      const int grow = m0 + rowW + mf * 16 + 8 * half + r;  // C: VGPR r -> M=r+8*half
      int e = 0;
#pragma unroll
      for (int ee = 0; ee < E_; ++ee) if (grow >= cum[ee + 1]) e = ee + 1;
#pragma unroll
      for (int nf = 0; nf < 4; ++nf) {
        const int gcol = n0 + colW + nf * 16 + l16;
        const float v  = acc[mf][nf][r] + B1[e * H_ + gcol];
        const float g  = 0.5f * v * (1.0f + erff(v * 0.70710678118654752f));
        H2U hi, lo;
        hi.h = (_Float16)g;
        lo.h = (_Float16)(g - (float)hi.h);
        Hout[(size_t)grow * H_ + gcol] = (uint32_t)hi.u | ((uint32_t)lo.u << 16);
      }
    }
  }
}

// ---------------------------------------------------------------------------
// Kernel 2: out = h @ W2[e]^T + b2[e].  A (h) comes pre-split from workspace.
// grid = (D/BN, N/BM, 2 batches), block = 256 threads.
// ---------------------------------------------------------------------------
__global__ __launch_bounds__(256)
void moe_gemm2(const uint32_t* __restrict__ Hbuf,
               const float* __restrict__ W2, const float* __restrict__ B2,
               const int*   __restrict__ counts,
               float*       __restrict__ Out)
{
  __shared__ _Float16 As[2][BM][LSTR];
  __shared__ _Float16 Bs[2][BN][LSTR];

  const int batch      = blockIdx.z;
  const uint32_t* Hin  = Hbuf + (size_t)batch * N_ * H_;
  float* O             = Out  + (size_t)batch * N_ * D_;

  const int m0 = blockIdx.y * BM;
  const int n0 = blockIdx.x * BN;

  int cum[E_ + 1];
  cum[0] = 0;
#pragma unroll
  for (int e = 0; e < E_; ++e) cum[e + 1] = cum[e] + counts[e];

  const int tid  = threadIdx.x;
  const int lane = tid & 31;
  const int wave = tid >> 5;
  const int half = lane >> 4;
  const int l16  = lane & 15;
  const int rowW = (wave >> 1) * 32;
  const int colW = (wave & 1)  * 64;

  v8f acc[2][4];
#pragma unroll
  for (int i = 0; i < 2; ++i)
#pragma unroll
    for (int j = 0; j < 4; ++j) acc[i][j] = v8f{};

  for (int e = 0; e < E_; ++e) {
    const int rs = max(m0, cum[e]);
    const int re = min(m0 + BM, cum[e + 1]);
    if (rs >= re) continue;
    const float* We = W2 + (size_t)e * D_ * H_;

    // ---- register-stage prefetch of tile k0 = 0 ----
    uint4  ra[4];
    float4 rb[4];
#pragma unroll
    for (int it = 0; it < 4; ++it) {
      int r, c; slot_rc(tid, it, r, c);
      const int grow = m0 + r;
      uint4 av = make_uint4(0u, 0u, 0u, 0u);
      if (grow >= rs && grow < re)
        av = *reinterpret_cast<const uint4*>(Hin + (size_t)grow * H_ + c);
      ra[it] = av;
      rb[it] = *reinterpret_cast<const float4*>(We + (size_t)(n0 + r) * H_ + c);
    }

    for (int k0 = 0; k0 < H_; k0 += BK) {
      __syncthreads();
      // ---- drain prefetch registers into LDS ----
#pragma unroll
      for (int it = 0; it < 4; ++it) {
        int r, c; slot_rc(tid, it, r, c);
        const uint32_t p[4] = {ra[it].x, ra[it].y, ra[it].z, ra[it].w};
        uint2 hp, lp;
        hp.x = (p[0] & 0xFFFFu) | ((p[1] & 0xFFFFu) << 16);
        hp.y = (p[2] & 0xFFFFu) | ((p[3] & 0xFFFFu) << 16);
        lp.x = (p[0] >> 16)     | ((p[1] >> 16) << 16);
        lp.y = (p[2] >> 16)     | ((p[3] >> 16) << 16);
        *reinterpret_cast<uint2*>(&As[0][r][c]) = hp;
        *reinterpret_cast<uint2*>(&As[1][r][c]) = lp;
        store_split(&Bs[0][r][c], &Bs[1][r][c], rb[it]);
      }
      __syncthreads();

      // ---- issue next tile's global loads ----
      if (k0 + BK < H_) {
        const int kn = k0 + BK;
#pragma unroll
        for (int it = 0; it < 4; ++it) {
          int r, c; slot_rc(tid, it, r, c);
          const int grow = m0 + r;
          uint4 av = make_uint4(0u, 0u, 0u, 0u);
          if (grow >= rs && grow < re)
            av = *reinterpret_cast<const uint4*>(Hin + (size_t)grow * H_ + kn + c);
          ra[it] = av;
          rb[it] = *reinterpret_cast<const float4*>(We + (size_t)(n0 + r) * H_ + kn + c);
        }
      }

      F16Frag aH[2], aL[2];
#pragma unroll
      for (int mf = 0; mf < 2; ++mf) {
        const _Float16* ah = &As[0][rowW + mf * 16 + l16][0];
        const _Float16* al = &As[1][rowW + mf * 16 + l16][0];
        aH[mf].q[0] = *reinterpret_cast<const uint4*>(ah + 8 * half);
        aH[mf].q[1] = *reinterpret_cast<const uint4*>(ah + 16 + 8 * half);
        aL[mf].q[0] = *reinterpret_cast<const uint4*>(al + 8 * half);
        aL[mf].q[1] = *reinterpret_cast<const uint4*>(al + 16 + 8 * half);
      }
#pragma unroll
      for (int nf = 0; nf < 4; ++nf) {
        F16Frag bH, bL;
        const _Float16* bh = &Bs[0][colW + nf * 16 + l16][0];
        const _Float16* bl = &Bs[1][colW + nf * 16 + l16][0];
        bH.q[0] = *reinterpret_cast<const uint4*>(bh + 8 * half);
        bH.q[1] = *reinterpret_cast<const uint4*>(bh + 16 + 8 * half);
        bL.q[0] = *reinterpret_cast<const uint4*>(bl + 8 * half);
        bL.q[1] = *reinterpret_cast<const uint4*>(bl + 16 + 8 * half);
#pragma unroll
        for (int mf = 0; mf < 2; ++mf)
          acc[mf][nf] = wmma3(aH[mf].v, aL[mf].v, bH.v, bL.v, acc[mf][nf]);
      }
    }
    __syncthreads();
  }

#pragma unroll
  for (int mf = 0; mf < 2; ++mf) {
#pragma unroll
    for (int r = 0; r < 8; ++r) {
      const int grow = m0 + rowW + mf * 16 + 8 * half + r;
      int e = 0;
#pragma unroll
      for (int ee = 0; ee < E_; ++ee) if (grow >= cum[ee + 1]) e = ee + 1;
#pragma unroll
      for (int nf = 0; nf < 4; ++nf) {
        const int gcol = n0 + colW + nf * 16 + l16;
        O[(size_t)grow * D_ + gcol] = acc[mf][nf][r] + B2[e * D_ + gcol];
      }
    }
  }
}

// ---------------------------------------------------------------------------
extern "C" void kernel_launch(void* const* d_in, const int* in_sizes, int n_in,
                              void* d_out, int out_size, void* d_ws, size_t ws_size,
                              hipStream_t stream) {
  const float* x      = (const float*)d_in[0];
  const float* y      = (const float*)d_in[1];
  const float* w1     = (const float*)d_in[2];
  const float* b1     = (const float*)d_in[3];
  const float* w2     = (const float*)d_in[4];
  const float* b2     = (const float*)d_in[5];
  const int*   counts = (const int*)d_in[6];
  float*       out    = (float*)d_out;
  uint32_t*    hbuf   = (uint32_t*)d_ws;   // 2 * N * H * 4B = 128 MB packed f16 pairs

  dim3 blk(256);
  dim3 g1(H_ / BN, N_ / BM, 2);
  moe_gemm1<<<g1, blk, 0, stream>>>(x, y, w1, b1, counts, hbuf);
  dim3 g2(D_ / BN, N_ / BM, 2);
  moe_gemm2<<<g2, blk, 0, stream>>>(hbuf, w2, b2, counts, out);
}
